// Net_knn_26396869001917
// MI455X (gfx1250) — compile-verified, matrix operand order checked by reading, not run on hardware
//
#include <hip/hip_runtime.h>

#define B_DIM  512
#define NB_DIM 43
#define F_DIM  512
#define HD_DIM 8192
#define NI_DIM 4096

#define BM 64
#define BN 128
#define BK  32          // fp32 fallback path K-step
#define BKT 64          // bf16/TDM path K-step (two WMMA k-slices per stage)
#define LDKF (BK + 4)   // fp32-path LDS pitch (bf16 elements, even)

typedef __attribute__((ext_vector_type(16))) __bf16   v16bf;
typedef __attribute__((ext_vector_type(8)))  float    v8f;
typedef unsigned __attribute__((ext_vector_type(4)))  u32x4;
typedef int      __attribute__((ext_vector_type(4)))  i32x4;
typedef int      __attribute__((ext_vector_type(8)))  i32x8;

#if defined(__has_builtin)
#if __has_builtin(__builtin_amdgcn_tensor_load_to_lds) && \
    __has_builtin(__builtin_amdgcn_s_wait_tensorcnt)
#define HAVE_TDM 1
#endif
#endif
#ifndef HAVE_TDM
#define HAVE_TDM 0
#endif

// ---------------------------------------------------------------- helpers ---
__device__ __forceinline__ unsigned short bfbits(float f) {
  union { float f; unsigned u; } c; c.f = f;
  unsigned r = c.u + 0x7FFFu + ((c.u >> 16) & 1u);  // round-to-nearest-even
  return (unsigned short)(r >> 16);
}
__device__ __forceinline__ unsigned pack2(float lo, float hi) {
  return ((unsigned)bfbits(hi) << 16) | (unsigned)bfbits(lo);
}
__device__ __forceinline__ void st_bf4(__bf16* d, float4 v) {
  unsigned* p = (unsigned*)d;
  p[0] = pack2(v.x, v.y);
  p[1] = pack2(v.z, v.w);
}

union Frag16 { v16bf v; unsigned u[8]; };

// Load one 16x32 bf16 fragment from an LDS tile stored row-major, K contiguous.
// CDNA5 16-bit matrix VGPR layout:
//   lanes 0-15 : VGPR j -> K=2j,2j+1     ; VGPR 4+j -> K=16+2j,16+2j+1
//   lanes 16-31: VGPR j -> K=8+2j,8+2j+1 ; VGPR 4+j -> K=24+2j,24+2j+1
__device__ __forceinline__ v16bf ld_frag(const __bf16* tile, int row, int hl, int pitch) {
  Frag16 f;
  const __bf16* p = tile + row * pitch + hl * 8;
#pragma unroll
  for (int j = 0; j < 4; ++j) {
    f.u[j]     = *(const unsigned*)(p + 2 * j);
    f.u[4 + j] = *(const unsigned*)(p + 16 + 2 * j);
  }
  return f.v;
}

// -------------------------------------------------- TDM 2D tile -> LDS ------
__device__ __forceinline__ void tdm_load_bf16(unsigned lds_byte_addr,
                                              const __bf16* tile_origin,
                                              unsigned tensor_dim0,   // row len (elems)
                                              unsigned tensor_dim1,   // #rows
                                              unsigned stride0,       // elems between rows
                                              unsigned tile_d0,       // tile K extent
                                              unsigned tile_d1) {     // tile rows
#if HAVE_TDM
  unsigned long long ga = (unsigned long long)tile_origin;
  u32x4 g0;
  g0.x = 1u;                                            // count=1, user-mode D#
  g0.y = lds_byte_addr;                                 // lds_addr  (bits 63:32)
  g0.z = (unsigned)(ga & 0xFFFFFFFFu);                  // global_addr[31:0]
  g0.w = (unsigned)((ga >> 32) & 0x01FFFFFFu)           // global_addr[56:32]
         | (2u << 30);                                  // type = 2 ("image")
  i32x8 g1;
  g1[0] = (int)(1u << 16);                              // data_size=1 -> 2 bytes
  g1[1] = (int)((tensor_dim0 & 0xFFFFu) << 16);         // tensor_dim0[15:0] @63:48
  g1[2] = (int)((tensor_dim0 >> 16) |
                ((tensor_dim1 & 0xFFFFu) << 16));       // tensor_dim1[15:0]
  g1[3] = (int)((tensor_dim1 >> 16) |
                (tile_d0 << 16));                       // tile_dim0 @127:112
  g1[4] = (int)(tile_d1 & 0xFFFFu);                     // tile_dim1; tile_dim2=0
  g1[5] = (int)stride0;                                 // tensor_dim0_stride[31:0]
  g1[6] = 0;
  g1[7] = 0;
  i32x4 z4 = {0, 0, 0, 0};
#if __clang_major__ >= 23
  i32x8 z8 = {0, 0, 0, 0, 0, 0, 0, 0};
  __builtin_amdgcn_tensor_load_to_lds(g0, g1, z4, z4, z8, 0);
#else
  __builtin_amdgcn_tensor_load_to_lds(g0, g1, z4, z4, 0);
#endif
#else
  (void)lds_byte_addr; (void)tile_origin; (void)tensor_dim0; (void)tensor_dim1;
  (void)stride0; (void)tile_d0; (void)tile_d1;
#endif
}

// Stage one (A,B) tile pair into LDS buffers: TDM when available (wave 0
// issues, no wait here -> overlaps with compute), manual copies otherwise.
__device__ __forceinline__ void stage_tiles(int tid, __bf16* Asb, __bf16* Bsb,
                                            const __bf16* aorg, unsigned adim0,
                                            unsigned adim1, unsigned astr,
                                            const __bf16* borg, unsigned bdim0,
                                            unsigned bdim1, unsigned bstr) {
#if HAVE_TDM
  if (tid == 0) {
    tdm_load_bf16((unsigned)(unsigned long long)Asb, aorg, adim0, adim1, astr, BKT, BM);
    tdm_load_bf16((unsigned)(unsigned long long)Bsb, borg, bdim0, bdim1, bstr, BKT, BN);
  }
#else
  const int r0 = tid >> 3, c0 = (tid & 7) * 8;
  const int r1 = (tid + 256) >> 3, c1 = ((tid + 256) & 7) * 8;
  uint4 a0 = *(const uint4*)(aorg + (size_t)r0 * astr + c0);
  uint4 a1 = *(const uint4*)(aorg + (size_t)r1 * astr + c1);
  uint4 b0 = *(const uint4*)(borg + (size_t)r0 * bstr + c0);
  uint4 b1 = *(const uint4*)(borg + (size_t)r1 * bstr + c1);
  uint4 b2 = *(const uint4*)(borg + (size_t)((tid + 512) >> 3) * bstr + (((tid + 512) & 7) * 8));
  uint4 b3 = *(const uint4*)(borg + (size_t)((tid + 768) >> 3) * bstr + (((tid + 768) & 7) * 8));
  *(uint4*)(Asb + r0 * BKT + c0) = a0;
  *(uint4*)(Asb + r1 * BKT + c1) = a1;
  *(uint4*)(Bsb + r0 * BKT + c0) = b0;
  *(uint4*)(Bsb + r1 * BKT + c1) = b1;
  *(uint4*)(Bsb + ((tid + 512) >> 3) * BKT + (((tid + 512) & 7) * 8)) = b2;
  *(uint4*)(Bsb + ((tid + 768) >> 3) * BKT + (((tid + 768) & 7) * 8)) = b3;
  (void)adim0; (void)adim1; (void)bdim0; (void)bdim1;
#endif
}

// Compute a 64-wide K slab: 2 k-slices x (2x2) WMMAs from one buffer pair.
__device__ __forceinline__ void compute_slab(const __bf16* Asb, const __bf16* Bsb,
                                             int waveM, int waveN, int l15, int hl,
                                             v8f acc[2][2]) {
#pragma unroll
  for (int ks = 0; ks < 2; ++ks) {
    v16bf af[2], bfr[2];
#pragma unroll
    for (int mi = 0; mi < 2; ++mi)
      af[mi] = ld_frag(Asb + (waveM * 32 + mi * 16) * BKT + ks * 32, l15, hl, BKT);
#pragma unroll
    for (int ni = 0; ni < 2; ++ni)
      bfr[ni] = ld_frag(Bsb + (waveN * 32 + ni * 16) * BKT + ks * 32, l15, hl, BKT);
#pragma unroll
    for (int mi = 0; mi < 2; ++mi)
#pragma unroll
      for (int ni = 0; ni < 2; ++ni)
        acc[mi][ni] = __builtin_amdgcn_wmma_f32_16x16x32_bf16(
            false, af[mi], false, bfr[ni], (short)0, acc[mi][ni], false, false);
  }
}

// --------------------------------------------- fp32 -> bf16 bulk convert ----
__global__ __launch_bounds__(256)
void cvt_bf16_kernel(const float* __restrict__ s, __bf16* __restrict__ d) {
  const size_t i = ((size_t)blockIdx.x * 256 + threadIdx.x) * 4;
  const float4 v = *(const float4*)(s + i);
  unsigned* p = (unsigned*)(d + i);
  p[0] = pack2(v.x, v.y);
  p[1] = pack2(v.z, v.w);
}

// ===========================================================================
// bf16-native path: TDM double-buffered software pipeline. DMA for stage p+1
// overlaps the 8 WMMAs of stage p; s_wait_tensorcnt only at the stage top.
// ===========================================================================
__global__ __launch_bounds__(256)
void gemm1_sign_bf(const __bf16* __restrict__ xb,    // [B, NB, F] bf16
                   const __bf16* __restrict__ w1b,   // [HD, F]    bf16
                   const float*  __restrict__ bp,    // [NB, HD]
                   float*        __restrict__ x1b,   // [B, HD] fp32 out
                   __bf16*       __restrict__ x1bb) {// [B, HD] bf16 out (ws)
  __shared__ __bf16 As[2][BM * BKT];
  __shared__ __bf16 Bs[2][BN * BKT];

  const int tid = threadIdx.x, lane = tid & 31, hl = lane >> 4, l15 = lane & 15;
  const int wave = tid >> 5, waveM = wave >> 2, waveN = wave & 3;
  const int nBase = blockIdx.x * BN;
  const int mBase = blockIdx.y * BM;

  const int T1   = F_DIM / BKT;        // 8 K-slabs per band
  const int PTOT = NB_DIM * T1;        // 344 pipeline stages

  float sums[2][2][8];
#pragma unroll
  for (int mi = 0; mi < 2; ++mi)
#pragma unroll
    for (int ni = 0; ni < 2; ++ni)
#pragma unroll
      for (int r = 0; r < 8; ++r) sums[mi][ni][r] = 0.0f;

  // Prologue: DMA stage 0 into buffer 0.
  stage_tiles(tid, As[0], Bs[0],
              xb + (size_t)mBase * (NB_DIM * F_DIM), NB_DIM * F_DIM, B_DIM, NB_DIM * F_DIM,
              w1b + (size_t)nBase * F_DIM, F_DIM, HD_DIM, F_DIM);

  v8f acc[2][2];
  for (int p = 0; p < PTOT; ++p) {
    const int band = p >> 3;           // T1 == 8
    const int ktm  = p & 7;
    if (ktm == 0) {
#pragma unroll
      for (int mi = 0; mi < 2; ++mi)
#pragma unroll
        for (int ni = 0; ni < 2; ++ni)
#pragma unroll
          for (int r = 0; r < 8; ++r) acc[mi][ni][r] = 0.0f;
    }
#if HAVE_TDM
    if (tid == 0) __builtin_amdgcn_s_wait_tensorcnt(0);  // stage p's DMA landed
#endif
    __syncthreads();

    if (p + 1 < PTOT) {                // prefetch stage p+1 into other buffer
      const int band2 = (p + 1) >> 3;
      const int k2    = ((p + 1) & 7) * BKT;
      stage_tiles(tid, As[(p + 1) & 1], Bs[(p + 1) & 1],
                  xb + (size_t)mBase * (NB_DIM * F_DIM) + (size_t)band2 * F_DIM + k2,
                  NB_DIM * F_DIM, B_DIM, NB_DIM * F_DIM,
                  w1b + (size_t)nBase * F_DIM + k2, F_DIM, HD_DIM, F_DIM);
    }

    compute_slab(As[p & 1], Bs[p & 1], waveM, waveN, l15, hl, acc);

    if (ktm == 7) {                    // band complete: sum += bp * sign(x1)
#pragma unroll
      for (int ni = 0; ni < 2; ++ni) {
        const int col  = nBase + waveN * 32 + ni * 16 + l15;
        const float bv = bp[(size_t)band * HD_DIM + col];
#pragma unroll
        for (int mi = 0; mi < 2; ++mi)
#pragma unroll
          for (int r = 0; r < 8; ++r)
            sums[mi][ni][r] += (acc[mi][ni][r] > 0.0f) ? bv : -bv;
      }
    }
    __syncthreads();
  }

#pragma unroll
  for (int mi = 0; mi < 2; ++mi) {
    const int rowb = mBase + waveM * 32 + mi * 16 + hl * 8;
#pragma unroll
    for (int ni = 0; ni < 2; ++ni) {
      const int col = nBase + waveN * 32 + ni * 16 + l15;
#pragma unroll
      for (int r = 0; r < 8; ++r) {
        const bool pos = sums[mi][ni][r] > 0.0f;
        x1b[(size_t)(rowb + r) * HD_DIM + col]  = pos ? 1.0f : -1.0f;
        x1bb[(size_t)(rowb + r) * HD_DIM + col] =
            __builtin_bit_cast(__bf16, (unsigned short)(pos ? 0x3F80u : 0xBF80u));
      }
    }
  }
}

__global__ __launch_bounds__(256)
void gemm2_bf(const __bf16* __restrict__ x1bb,  // [B, HD] bf16
              const __bf16* __restrict__ w2b,   // [NI, HD] bf16
              float*        __restrict__ out) { // [B, NI]
  __shared__ __bf16 As[2][BM * BKT];
  __shared__ __bf16 Bs[2][BN * BKT];

  const int tid = threadIdx.x, lane = tid & 31, hl = lane >> 4, l15 = lane & 15;
  const int wave = tid >> 5, waveM = wave >> 2, waveN = wave & 3;
  const int nBase = blockIdx.x * BN;
  const int mBase = blockIdx.y * BM;
  const int T2 = HD_DIM / BKT;         // 128 stages

  v8f acc[2][2];
#pragma unroll
  for (int mi = 0; mi < 2; ++mi)
#pragma unroll
    for (int ni = 0; ni < 2; ++ni)
#pragma unroll
      for (int r = 0; r < 8; ++r) acc[mi][ni][r] = 0.0f;

  stage_tiles(tid, As[0], Bs[0],
              x1bb + (size_t)mBase * HD_DIM, HD_DIM, B_DIM, HD_DIM,
              w2b + (size_t)nBase * HD_DIM, HD_DIM, NI_DIM, HD_DIM);

  for (int kt = 0; kt < T2; ++kt) {
#if HAVE_TDM
    if (tid == 0) __builtin_amdgcn_s_wait_tensorcnt(0);
#endif
    __syncthreads();

    if (kt + 1 < T2) {
      const int k2 = (kt + 1) * BKT;
      stage_tiles(tid, As[(kt + 1) & 1], Bs[(kt + 1) & 1],
                  x1bb + (size_t)mBase * HD_DIM + k2, HD_DIM, B_DIM, HD_DIM,
                  w2b + (size_t)nBase * HD_DIM + k2, HD_DIM, NI_DIM, HD_DIM);
    }

    compute_slab(As[kt & 1], Bs[kt & 1], waveM, waveN, l15, hl, acc);
    __syncthreads();
  }

  const float inv = 1.0f / (float)HD_DIM;
#pragma unroll
  for (int mi = 0; mi < 2; ++mi) {
    const int rowb = mBase + waveM * 32 + mi * 16 + hl * 8;
#pragma unroll
    for (int ni = 0; ni < 2; ++ni) {
      const int col = nBase + waveN * 32 + ni * 16 + l15;
#pragma unroll
      for (int r = 0; r < 8; ++r)
        out[(size_t)(rowb + r) * NI_DIM + col] = acc[mi][ni][r] * inv;
    }
  }
}

// ===========================================================================
// Fallback path (workspace too small): fused fp32->bf16 conversion staging.
// ===========================================================================
__global__ __launch_bounds__(256)
void gemm1_sign_kernel(const float* __restrict__ x, const float* __restrict__ W1,
                       const float* __restrict__ bp, float* __restrict__ x1b) {
  __shared__ __bf16 As[BM * LDKF];
  __shared__ __bf16 Bs[BN * LDKF];

  const int tid = threadIdx.x, lane = tid & 31, hl = lane >> 4, l15 = lane & 15;
  const int wave = tid >> 5, waveM = wave >> 2, waveN = wave & 3;
  const int nBase = blockIdx.x * BN;
  const int mBase = blockIdx.y * BM;

  float sums[2][2][8];
#pragma unroll
  for (int mi = 0; mi < 2; ++mi)
#pragma unroll
    for (int ni = 0; ni < 2; ++ni)
#pragma unroll
      for (int r = 0; r < 8; ++r) sums[mi][ni][r] = 0.0f;

  for (int band = 0; band < NB_DIM; ++band) {
    v8f acc[2][2];
#pragma unroll
    for (int mi = 0; mi < 2; ++mi)
#pragma unroll
      for (int ni = 0; ni < 2; ++ni)
#pragma unroll
        for (int r = 0; r < 8; ++r) acc[mi][ni][r] = 0.0f;

    for (int kt = 0; kt < F_DIM / BK; ++kt) {
      const int k0 = kt * BK;
      const int ia0 = tid, ia1 = tid + 256;
      const float4 a0 = *(const float4*)(x + (size_t)(mBase + (ia0 >> 3)) * (NB_DIM * F_DIM)
                                           + (size_t)band * F_DIM + k0 + (ia0 & 7) * 4);
      const float4 a1 = *(const float4*)(x + (size_t)(mBase + (ia1 >> 3)) * (NB_DIM * F_DIM)
                                           + (size_t)band * F_DIM + k0 + (ia1 & 7) * 4);
      float4 bv[4];
#pragma unroll
      for (int j = 0; j < 4; ++j) {
        const int ib = tid + j * 256;
        bv[j] = *(const float4*)(W1 + (size_t)(nBase + (ib >> 3)) * F_DIM + k0 + (ib & 7) * 4);
      }
      st_bf4(&As[(ia0 >> 3) * LDKF + (ia0 & 7) * 4], a0);
      st_bf4(&As[(ia1 >> 3) * LDKF + (ia1 & 7) * 4], a1);
#pragma unroll
      for (int j = 0; j < 4; ++j) {
        const int ib = tid + j * 256;
        st_bf4(&Bs[(ib >> 3) * LDKF + (ib & 7) * 4], bv[j]);
      }
      __syncthreads();

      v16bf af[2], bfr[2];
#pragma unroll
      for (int mi = 0; mi < 2; ++mi)
        af[mi] = ld_frag(&As[(waveM * 32 + mi * 16) * LDKF], l15, hl, LDKF);
#pragma unroll
      for (int ni = 0; ni < 2; ++ni)
        bfr[ni] = ld_frag(&Bs[(waveN * 32 + ni * 16) * LDKF], l15, hl, LDKF);
#pragma unroll
      for (int mi = 0; mi < 2; ++mi)
#pragma unroll
        for (int ni = 0; ni < 2; ++ni)
          acc[mi][ni] = __builtin_amdgcn_wmma_f32_16x16x32_bf16(
              false, af[mi], false, bfr[ni], (short)0, acc[mi][ni], false, false);
      __syncthreads();
    }

#pragma unroll
    for (int ni = 0; ni < 2; ++ni) {
      const int col  = nBase + waveN * 32 + ni * 16 + l15;
      const float bv = bp[(size_t)band * HD_DIM + col];
#pragma unroll
      for (int mi = 0; mi < 2; ++mi)
#pragma unroll
        for (int r = 0; r < 8; ++r)
          sums[mi][ni][r] += (acc[mi][ni][r] > 0.0f) ? bv : -bv;
    }
  }

#pragma unroll
  for (int mi = 0; mi < 2; ++mi) {
    const int rowb = mBase + waveM * 32 + mi * 16 + hl * 8;
#pragma unroll
    for (int ni = 0; ni < 2; ++ni) {
      const int col = nBase + waveN * 32 + ni * 16 + l15;
#pragma unroll
      for (int r = 0; r < 8; ++r)
        x1b[(size_t)(rowb + r) * HD_DIM + col] = (sums[mi][ni][r] > 0.0f) ? 1.0f : -1.0f;
    }
  }
}

__global__ __launch_bounds__(256)
void gemm2_kernel(const float* __restrict__ x1b, const float* __restrict__ W2,
                  float* __restrict__ out) {
  __shared__ __bf16 As[BM * LDKF];
  __shared__ __bf16 Bs[BN * LDKF];

  const int tid = threadIdx.x, lane = tid & 31, hl = lane >> 4, l15 = lane & 15;
  const int wave = tid >> 5, waveM = wave >> 2, waveN = wave & 3;
  const int nBase = blockIdx.x * BN;
  const int mBase = blockIdx.y * BM;

  v8f acc[2][2];
#pragma unroll
  for (int mi = 0; mi < 2; ++mi)
#pragma unroll
    for (int ni = 0; ni < 2; ++ni)
#pragma unroll
      for (int r = 0; r < 8; ++r) acc[mi][ni][r] = 0.0f;

  for (int kt = 0; kt < HD_DIM / BK; ++kt) {
    const int k0 = kt * BK;
    const int ia0 = tid, ia1 = tid + 256;
    const float4 a0 = *(const float4*)(x1b + (size_t)(mBase + (ia0 >> 3)) * HD_DIM + k0 + (ia0 & 7) * 4);
    const float4 a1 = *(const float4*)(x1b + (size_t)(mBase + (ia1 >> 3)) * HD_DIM + k0 + (ia1 & 7) * 4);
    float4 bv[4];
#pragma unroll
    for (int j = 0; j < 4; ++j) {
      const int ib = tid + j * 256;
      bv[j] = *(const float4*)(W2 + (size_t)(nBase + (ib >> 3)) * HD_DIM + k0 + (ib & 7) * 4);
    }
    st_bf4(&As[(ia0 >> 3) * LDKF + (ia0 & 7) * 4], a0);
    st_bf4(&As[(ia1 >> 3) * LDKF + (ia1 & 7) * 4], a1);
#pragma unroll
    for (int j = 0; j < 4; ++j) {
      const int ib = tid + j * 256;
      st_bf4(&Bs[(ib >> 3) * LDKF + (ib & 7) * 4], bv[j]);
    }
    __syncthreads();

    v16bf af[2], bfr[2];
#pragma unroll
    for (int mi = 0; mi < 2; ++mi)
      af[mi] = ld_frag(&As[(waveM * 32 + mi * 16) * LDKF], l15, hl, LDKF);
#pragma unroll
    for (int ni = 0; ni < 2; ++ni)
      bfr[ni] = ld_frag(&Bs[(waveN * 32 + ni * 16) * LDKF], l15, hl, LDKF);
#pragma unroll
    for (int mi = 0; mi < 2; ++mi)
#pragma unroll
      for (int ni = 0; ni < 2; ++ni)
        acc[mi][ni] = __builtin_amdgcn_wmma_f32_16x16x32_bf16(
            false, af[mi], false, bfr[ni], (short)0, acc[mi][ni], false, false);
    __syncthreads();
  }

  const float inv = 1.0f / (float)HD_DIM;
#pragma unroll
  for (int mi = 0; mi < 2; ++mi) {
    const int rowb = mBase + waveM * 32 + mi * 16 + hl * 8;
#pragma unroll
    for (int ni = 0; ni < 2; ++ni) {
      const int col = nBase + waveN * 32 + ni * 16 + l15;
#pragma unroll
      for (int r = 0; r < 8; ++r)
        out[(size_t)(rowb + r) * NI_DIM + col] = acc[mi][ni][r] * inv;
    }
  }
}

// ===========================================================================
extern "C" void kernel_launch(void* const* d_in, const int* in_sizes, int n_in,
                              void* d_out, int out_size, void* d_ws, size_t ws_size,
                              hipStream_t stream) {
  (void)in_sizes; (void)n_in; (void)out_size;
  const float* x  = (const float*)d_in[0];   // [512, 43, 512]
  const float* W1 = (const float*)d_in[1];   // [8192, 512]
  const float* W2 = (const float*)d_in[2];   // [4096, 8192]
  const float* bp = (const float*)d_in[3];   // [43, 8192]

  float* out = (float*)d_out;                        // [512, 4096]
  float* x1b = out + (size_t)B_DIM * NI_DIM;         // [512, 8192]

  const size_t nx  = (size_t)B_DIM * NB_DIM * F_DIM;
  const size_t nw1 = (size_t)HD_DIM * F_DIM;
  const size_t nw2 = (size_t)NI_DIM * HD_DIM;
  const size_t nxb = (size_t)B_DIM * HD_DIM;
  const size_t need = (nx + nw1 + nw2 + nxb) * sizeof(__bf16);

  dim3 blk(256);
  if (ws_size >= need) {
    __bf16* xb   = (__bf16*)d_ws;
    __bf16* w1b  = xb + nx;
    __bf16* w2b  = w1b + nw1;
    __bf16* x1bb = w2b + nw2;
    cvt_bf16_kernel<<<(unsigned)(nx  / 1024), blk, 0, stream>>>(x,  xb);
    cvt_bf16_kernel<<<(unsigned)(nw1 / 1024), blk, 0, stream>>>(W1, w1b);
    cvt_bf16_kernel<<<(unsigned)(nw2 / 1024), blk, 0, stream>>>(W2, w2b);

    dim3 g1(HD_DIM / BN, B_DIM / BM);   // 64 x 8
    gemm1_sign_bf<<<g1, blk, 0, stream>>>(xb, w1b, bp, x1b, x1bb);
    dim3 g2(NI_DIM / BN, B_DIM / BM);   // 32 x 8
    gemm2_bf<<<g2, blk, 0, stream>>>(x1bb, w2b, out);
  } else {
    dim3 g1(HD_DIM / BN, B_DIM / BM);
    gemm1_sign_kernel<<<g1, blk, 0, stream>>>(x, W1, bp, x1b);
    dim3 g2(NI_DIM / BN, B_DIM / BM);
    gemm2_kernel<<<g2, blk, 0, stream>>>(x1b, W2, out);
  }
}
